// Encoder_32538672235167
// MI455X (gfx1250) — compile-verified, matrix operand order checked by reading, not run on hardware
//
#include <hip/hip_runtime.h>
#include <hip/hip_bf16.h>

// LSTM encoder: B=32, T=2048, E=512, H=512.
// Kernel 1: one-time f32->bf16 transpose of src_seq into [T,B,E] (bandwidth
//           bound, ~9us at 23.3 TB/s).
// Kernel 2: 16 persistent workgroups (one H-slice each), W/U slices resident
//           in 320KB LDS as bf16, per-step GEMMs via v_wmma_f32_16x16x32_bf16,
//           one grid barrier per timestep via monotonic counter in d_ws.

#define B_   32
#define T_   2048
#define E_   512
#define H_   512
#define G4_  (4 * H_)
#define NWG  16
#define LDSK 520   // padded K stride (bf16 elems): 1040B/col -> conflict-free, 16B aligned
#define ZPAD 36    // padded row stride (floats) for z staging: 144B/col, 16B aligned

typedef __attribute__((ext_vector_type(16))) __bf16 v16bf;
typedef __attribute__((ext_vector_type(8)))  float  v8f;

union BfChunk { uint4 u; __bf16 h[8]; };

// Load 8 consecutive bf16 (16B, aligned) into fragment positions [base..base+7]
__device__ __forceinline__ void load8_bf(const __bf16* p, v16bf& f, int base) {
  BfChunk c;
  c.u = *(const uint4*)p;
#pragma unroll
  for (int i = 0; i < 8; ++i) f[base + i] = c.h[i];
}

__device__ __forceinline__ float sigmoidf_(float x) {
  return 1.0f / (1.0f + __expf(-x));
}

// Reset barrier counter; seed h buffer 0 with bf16(h0).
__global__ void lstm_init(const float* __restrict__ h0, __bf16* __restrict__ hb0,
                          int* __restrict__ cnt) {
  int i = blockIdx.x * blockDim.x + threadIdx.x;
  if (i == 0) *cnt = 0;
  if (i < B_ * H_) hb0[i] = (__bf16)h0[i];
}

// src [B,T,E] f32  ->  xb [T,B,E] bf16 (one-time, fully parallel, coalesced)
__global__ void __launch_bounds__(256)
lstm_cvt_x(const float* __restrict__ src, __bf16* __restrict__ xb) {
  // one thread = 8 consecutive E elements
  size_t i = (size_t)blockIdx.x * blockDim.x + threadIdx.x;  // < B*T*E/8
  int e8 = (int)(i & 63);          // E/8 = 64 chunks
  int b  = (int)((i >> 6) & 31);
  int t  = (int)(i >> 11);
  const float* p = src + ((size_t)b * T_ + t) * E_ + e8 * 8;
  float4 a = *(const float4*)p;
  float4 c = *(const float4*)(p + 4);
  BfChunk o;
  o.h[0] = (__bf16)a.x; o.h[1] = (__bf16)a.y; o.h[2] = (__bf16)a.z; o.h[3] = (__bf16)a.w;
  o.h[4] = (__bf16)c.x; o.h[5] = (__bf16)c.y; o.h[6] = (__bf16)c.z; o.h[7] = (__bf16)c.w;
  *(uint4*)(xb + ((size_t)t * B_ + b) * E_ + e8 * 8) = o.u;
}

__global__ void __launch_bounds__(256, 1)
lstm_persistent(const __bf16* __restrict__ xb,   // [T,B,E] bf16
                const float* __restrict__ W,     // [E,4H] row-major
                const float* __restrict__ U,     // [H,4H] row-major
                const float* __restrict__ bias,  // [4H]
                const float* __restrict__ c0,    // [B,H]
                float* __restrict__ out,         // h_T | hs[T,B,H] | h_T
                __bf16* __restrict__ hb0,        // [B,H] bf16 double buffer
                __bf16* __restrict__ hb1,
                int* __restrict__ cnt) {
  extern __shared__ char smem[];
  __bf16* Wl = (__bf16*)smem;             // [128][LDSK] column-major slice of W
  __bf16* Ul = Wl + 128 * LDSK;           // [128][LDSK] column-major slice of U
  float*  zb = (float*)(Ul + 128 * LDSK); // [128 cols][ZPAD rows] z staging

  const int wg    = blockIdx.x;           // owns h columns [wg*32, wg*32+32)
  const int tid   = threadIdx.x;
  const int wave  = tid >> 5;             // 8 waves: wave owns local cols [wave*16, +16)
  const int lane  = tid & 31;
  const int lhalf = lane >> 4;            // ISA 16-bit fragment half select
  const int l16   = lane & 15;
  const int kb    = lhalf ? 8 : 0;        // K sub-offset per lane half (ISA 7.12.2)

  // ---- one-time: stage weight slices into LDS (f32 -> bf16, col-major, padded) ----
  for (int idx = tid; idx < 128 * E_; idx += 256) {
    int c = idx & 127;                    // local col: g*32 + hc
    int k = idx >> 7;
    int g = c >> 5;
    int gcol = g * H_ + wg * 32 + (c & 31);
    Wl[c * LDSK + k] = (__bf16)W[(size_t)k * G4_ + gcol];
    Ul[c * LDSK + k] = (__bf16)U[(size_t)k * G4_ + gcol];
  }

  // ---- per-thread elementwise ownership: 4 (row,hcol) cells, c in registers ----
  float c_reg[4], h_reg[4], bi_[4], bf_[4], bg_[4], bo_[4];
  int   erow[4], ecol[4];
#pragma unroll
  for (int q = 0; q < 4; ++q) {
    int e = tid * 4 + q;
    int row = e >> 5, hc = e & 31;
    int wcol = wg * 32 + hc;
    erow[q] = row; ecol[q] = wcol;
    c_reg[q] = c0[row * H_ + wcol];
    h_reg[q] = 0.0f;
    bi_[q] = bias[0 * H_ + wcol];
    bf_[q] = bias[1 * H_ + wcol];
    bg_[q] = bias[2 * H_ + wcol];
    bo_[q] = bias[3 * H_ + wcol];
  }
  __syncthreads();

  const int n0 = wave * 16;               // local column tile base

  for (int t = 0; t < T_; ++t) {
    const __bf16* hcur = (t & 1) ? hb1 : hb0;
    __bf16*       hnxt = (t & 1) ? hb0 : hb1;
    const __bf16* xt   = xb + (size_t)t * B_ * E_;   // contiguous 32KB tile

    v8f acc[2];
#pragma unroll
    for (int m = 0; m < 2; ++m)
#pragma unroll
      for (int r = 0; r < 8; ++r) acc[m][r] = 0.0f;

    // ---- z = x_t @ W_slice + h_t @ U_slice  (bf16 WMMA, fp32 accum) ----
#pragma unroll 2
    for (int k0 = 0; k0 < E_; k0 += 32) {
      v16bf bw, bu;
      {
        const __bf16* pw = Wl + (n0 + l16) * LDSK + k0 + kb;
        load8_bf(pw,      bw, 0);
        load8_bf(pw + 16, bw, 8);
        const __bf16* pu = Ul + (n0 + l16) * LDSK + k0 + kb;
        load8_bf(pu,      bu, 0);
        load8_bf(pu + 16, bu, 8);
      }
#pragma unroll
      for (int m = 0; m < 2; ++m) {
        int batch = m * 16 + l16;         // A-matrix row per lane
        v16bf ax, ah;
        const __bf16* px = xt + batch * E_ + k0 + kb;
        load8_bf(px,      ax, 0);
        load8_bf(px + 16, ax, 8);
        const __bf16* ph = hcur + batch * H_ + k0 + kb;
        load8_bf(ph,      ah, 0);
        load8_bf(ph + 16, ah, 8);
        acc[m] = __builtin_amdgcn_wmma_f32_16x16x32_bf16(
            false, ax, false, bw, (short)0, acc[m], false, false);
        acc[m] = __builtin_amdgcn_wmma_f32_16x16x32_bf16(
            false, ah, false, bu, (short)0, acc[m], false, false);
      }
    }

    // ---- stage z tiles to LDS (C layout: VGPR r -> row lhalf*8+r, col l16) ----
    {
      int col = n0 + l16;
#pragma unroll
      for (int m = 0; m < 2; ++m) {
        float* p = zb + col * ZPAD + m * 16 + lhalf * 8;
        *(float4*)(p + 0) = make_float4(acc[m][0], acc[m][1], acc[m][2], acc[m][3]);
        *(float4*)(p + 4) = make_float4(acc[m][4], acc[m][5], acc[m][6], acc[m][7]);
      }
    }
    __syncthreads();

    // ---- gates + state update; write hs[t] and next-step bf16 h ----
#pragma unroll
    for (int q = 0; q < 4; ++q) {
      int row = erow[q];
      int hc  = ecol[q] - wg * 32;
      float zi = zb[(0 * 32 + hc) * ZPAD + row] + bi_[q];
      float zf = zb[(1 * 32 + hc) * ZPAD + row] + bf_[q];
      float zg = zb[(2 * 32 + hc) * ZPAD + row] + bg_[q];
      float zo = zb[(3 * 32 + hc) * ZPAD + row] + bo_[q];
      float ig = sigmoidf_(zi);
      float fg = sigmoidf_(zf);
      float gg = tanhf(zg);
      float og = sigmoidf_(zo);
      float cn = fg * c_reg[q] + ig * gg;
      c_reg[q] = cn;
      float hn = og * tanhf(cn);
      h_reg[q] = hn;
      size_t oidx = (size_t)row * H_ + ecol[q];
      out[(size_t)B_ * H_ + (size_t)t * B_ * H_ + oidx] = hn;  // hs[t]
      hnxt[oidx] = (__bf16)hn;
    }

    // ---- grid-wide barrier (monotonic counter; also guards zb reuse) ----
    __syncthreads();
    if (tid == 0) {
      __threadfence();
      __hip_atomic_fetch_add(cnt, 1, __ATOMIC_ACQ_REL, __HIP_MEMORY_SCOPE_AGENT);
      const int target = NWG * (t + 1);
      while (__hip_atomic_load(cnt, __ATOMIC_ACQUIRE, __HIP_MEMORY_SCOPE_AGENT) < target)
        __builtin_amdgcn_s_sleep(8);
    }
    __syncthreads();
    __threadfence();
  }

  // ---- final h_T (output slot 0 and slot 2) ----
#pragma unroll
  for (int q = 0; q < 4; ++q) {
    size_t oidx = (size_t)erow[q] * H_ + ecol[q];
    out[oidx] = h_reg[q];
    out[(size_t)B_ * H_ + (size_t)T_ * B_ * H_ + oidx] = h_reg[q];
  }
}

extern "C" void kernel_launch(void* const* d_in, const int* in_sizes, int n_in,
                              void* d_out, int out_size, void* d_ws, size_t ws_size,
                              hipStream_t stream) {
  (void)in_sizes; (void)n_in; (void)out_size; (void)ws_size;
  const float* src  = (const float*)d_in[0];  // [B,T,E]
  const float* W    = (const float*)d_in[1];  // [E,4H]
  const float* U    = (const float*)d_in[2];  // [H,4H]
  const float* bias = (const float*)d_in[3];  // [4H]
  const float* h0   = (const float*)d_in[4];  // [B,H]
  const float* c0   = (const float*)d_in[5];  // [B,H]
  float* out = (float*)d_out;

  char* ws = (char*)d_ws;
  int*     cnt = (int*)ws;                         // barrier counter
  __bf16*  hb0 = (__bf16*)(ws + 256);              // [B,H] bf16 h, buffer 0
  __bf16*  hb1 = hb0 + (size_t)B_ * H_;            // buffer 1
  __bf16*  xb  = (__bf16*)(ws + (1 << 17));        // [T,B,E] bf16, 64 MB

  const size_t shmem = (size_t)2 * 128 * LDSK * sizeof(__bf16) +
                       (size_t)128 * ZPAD * sizeof(float);       // 284,672 B < 320 KB

  hipFuncSetAttribute((const void*)lstm_persistent,
                      hipFuncAttributeMaxDynamicSharedMemorySize, (int)shmem);

  lstm_init<<<(B_ * H_ + 255) / 256, 256, 0, stream>>>(h0, hb0, cnt);
  lstm_cvt_x<<<(int)(((size_t)B_ * T_ * E_ / 8) / 256), 256, 0, stream>>>(src, xb);
  lstm_persistent<<<NWG, 256, shmem, stream>>>(xb, W, U, bias, c0, out,
                                               hb0, hb1, cnt);
}
